// ReconRLA_Encoder_4217657885149
// MI455X (gfx1250) — matrix operand order repaired and checked by reading, not compiled, not run on hardware
//
#include <hip/hip_runtime.h>
#include <hip/hip_bf16.h>

typedef __attribute__((ext_vector_type(16))) _Float16 v16h;
typedef __attribute__((ext_vector_type(8)))  _Float16 v8h;
typedef __attribute__((ext_vector_type(8)))  float    v8f;
typedef __attribute__((ext_vector_type(4)))  float    v4f;
typedef __attribute__((ext_vector_type(4)))  unsigned int v4u;
typedef __attribute__((ext_vector_type(8)))  int      v8i;
typedef __attribute__((ext_vector_type(4)))  int      v4i;

#define BN_INV 0.9999950000374997f   // 1/sqrt(1+1e-5)

#if defined(__has_builtin)
#if __has_builtin(__builtin_amdgcn_tensor_load_to_lds) && \
    __has_builtin(__builtin_amdgcn_s_wait_tensorcnt)
#define HAVE_TDM 1
#endif
#endif

// ---------------------------------------------------------------------------
// WMMA helpers (CDNA5 gfx1250, wave32, 16x16x32 f16 -> f32)
// ---------------------------------------------------------------------------
__device__ __forceinline__ v8f wmma_f16(v16h a, v16h b, v8f c) {
  return __builtin_amdgcn_wmma_f32_16x16x32_f16(false, a, false, b, (short)0, c,
                                                false, false);
}

// A fragment (16x32 f16) from row-major f16 tile in LDS; ld mult of 8, 16B aligned.
__device__ __forceinline__ v16h lds_a(const _Float16* base, int ld, int k0) {
  int lane = threadIdx.x & 31;
  const _Float16* p = base + (lane & 15) * ld + k0 + ((lane >> 4) << 3);
  v8h lo = *(const v8h*)p;           // ds_load_b128
  v8h hi = *(const v8h*)(p + 16);    // ds_load_b128
  return __builtin_shufflevector(lo, hi, 0, 1, 2, 3, 4, 5, 6, 7,
                                 8, 9, 10, 11, 12, 13, 14, 15);
}

// A fragment from global f32, UNCONDITIONAL (buffer zero-padded, LDX mult of 32).
template <int LDX>
__device__ __forceinline__ v16h g_a_f32u(const float* X, int m0, int k0) {
  int lane = threadIdx.x & 31;
  const float* xr = X + (size_t)(m0 + (lane & 15)) * LDX + k0 + ((lane >> 4) << 3);
  v4f x0 = *(const v4f*)(xr);
  v4f x1 = *(const v4f*)(xr + 4);
  v4f x2 = *(const v4f*)(xr + 16);
  v4f x3 = *(const v4f*)(xr + 20);
  v16h a;
#pragma unroll
  for (int h = 0; h < 4; ++h) {
    a[h]      = (_Float16)x0[h];
    a[h + 4]  = (_Float16)x1[h];
    a[h + 8]  = (_Float16)x2[h];
    a[h + 12] = (_Float16)x3[h];
  }
  return a;
}

// B fragment (32x16) from TRANSPOSED zero-padded f16 weights Wt[N][LDK]:
// per lane n = n0+(lane&15) fixed, k in two runs of 8 (8h, 16+8h) -> two 16B loads.
template <int LDK>
__device__ __forceinline__ v16h frag_bT(const _Float16* Wt, int k0, int n0) {
  int lane = threadIdx.x & 31;
  const _Float16* p = Wt + (size_t)(n0 + (lane & 15)) * LDK + k0 + ((lane >> 4) << 3);
  v8h lo = *(const v8h*)p;
  v8h hi = *(const v8h*)(p + 16);
  return __builtin_shufflevector(lo, hi, 0, 1, 2, 3, 4, 5, 6, 7,
                                 8, 9, 10, 11, 12, 13, 14, 15);
}

// ---------------------------------------------------------------------------
// fc0: [M,3] @ W[3,8] + BN + ReLU, output stride 32, pad cols zeroed
// ---------------------------------------------------------------------------
__global__ void fc0_kernel(const float* __restrict__ x, const float* __restrict__ W,
                           const float* __restrict__ g, const float* __restrict__ b,
                           float* __restrict__ y, int M) {
  int i = blockIdx.x * blockDim.x + threadIdx.x;
  if (i >= M) return;
  float a0 = x[i * 3 + 0], a1 = x[i * 3 + 1], a2 = x[i * 3 + 2];
#pragma unroll
  for (int c = 0; c < 8; ++c) {
    float v = a0 * W[c] + a1 * W[8 + c] + a2 * W[16 + c];
    v = v * (g[c] * BN_INV) + b[c];
    y[(size_t)i * 32 + c] = fmaxf(v, 0.0f);
  }
#pragma unroll
  for (int c = 8; c < 32; ++c) y[(size_t)i * 32 + c] = 0.0f;
}

// ---------------------------------------------------------------------------
// conv_bn GEMM, compile-time dims. X[M][K] zero-padded, Wt[NP][K] f16
// transposed zero-padded, Y[M][NP] (pad cols written 0).
// ---------------------------------------------------------------------------
template <int K, int NOUT, int NP>
__global__ void convbn_kernel(const float* __restrict__ X, const _Float16* __restrict__ Wt,
                              const float* __restrict__ g, const float* __restrict__ bv,
                              float* __restrict__ Y) {
  int m0 = blockIdx.x * 16;
  int n0 = blockIdx.y * 16;
  __builtin_prefetch(Wt, 0, 0);
  v8f c = {};
#pragma unroll
  for (int k0 = 0; k0 < K; k0 += 32)
    c = wmma_f16(g_a_f32u<K>(X, m0, k0), frag_bT<K>(Wt, k0, n0), c);
  int lane = threadIdx.x & 31;
  int nn = n0 + (lane & 15);
  bool ok = nn < NOUT;
  float s = ok ? g[nn] * BN_INV : 0.0f;
  float bb = ok ? bv[nn] : 0.0f;
  int mb = m0 + ((lane >> 4) << 3);
#pragma unroll
  for (int v = 0; v < 8; ++v)
    Y[(size_t)(mb + v) * NP + nn] = fmaxf(c[v] * s + bb, 0.0f);
}

// ---------------------------------------------------------------------------
// Fused residual: leaky_relu( bn(Fbb@W2) + bn(Fin@Wsc) ), NOUT mult of 16
// ---------------------------------------------------------------------------
template <int K1, int K2, int NOUT>
__global__ void resout_kernel(const float* __restrict__ Fbb, const _Float16* __restrict__ W2t,
                              const float* __restrict__ g2, const float* __restrict__ b2,
                              const float* __restrict__ Fin, const _Float16* __restrict__ Wsct,
                              const float* __restrict__ gsc, const float* __restrict__ bsc,
                              float* __restrict__ Y) {
  int m0 = blockIdx.x * 16;
  int n0 = blockIdx.y * 16;
  __builtin_prefetch(W2t, 0, 0);
  __builtin_prefetch(Wsct, 0, 0);
  v8f c1 = {}, c2 = {};
#pragma unroll
  for (int k0 = 0; k0 < K1; k0 += 32)
    c1 = wmma_f16(g_a_f32u<K1>(Fbb, m0, k0), frag_bT<K1>(W2t, k0, n0), c1);
#pragma unroll
  for (int k0 = 0; k0 < K2; k0 += 32)
    c2 = wmma_f16(g_a_f32u<K2>(Fin, m0, k0), frag_bT<K2>(Wsct, k0, n0), c2);
  int lane = threadIdx.x & 31;
  int nn = n0 + (lane & 15);
  float s2 = g2[nn] * BN_INV, bb2 = b2[nn];
  float ss = gsc[nn] * BN_INV, bbs = bsc[nn];
  int mb = m0 + ((lane >> 4) << 3);
#pragma unroll
  for (int v = 0; v < 8; ++v) {
    float y = c1[v] * s2 + bb2 + c2[v] * ss + bbs;
    y = (y > 0.0f) ? y : 0.2f * y;
    Y[(size_t)(mb + v) * NOUT + nn] = y;
  }
}

// ---------------------------------------------------------------------------
// Building-block kernel: one wave per point.
// Weights transposed/padded: bb1Wt[NPH][32], bb2Wt[NPH][FXW], fcWt[CD][XW].
// For CD<=64 the fcWt matrix is staged block-wide into LDS via the Tensor
// Data Mover (TENSOR_LOAD_TO_LDS + s_wait_tensorcnt).
// ---------------------------------------------------------------------------
template <int CD, bool PHASE2, int WPB>
__global__ __launch_bounds__(WPB * 32)
void bb_kernel(const float* __restrict__ xyz, const int* __restrict__ neigh,
               const float* __restrict__ fsrc,
               const _Float16* __restrict__ bb1W, const float* __restrict__ bb1g,
               const float* __restrict__ bb1b,
               const _Float16* __restrict__ bb2W, const float* __restrict__ bb2g,
               const float* __restrict__ bb2b,
               const _Float16* __restrict__ fcW,
               const float* __restrict__ mlpW,   // [CD][DOUTP] f32 exact
               const float* __restrict__ mlpg, const float* __restrict__ mlpb,
               float* __restrict__ out, int N) {
  constexpr int DH = CD / 2;
  constexpr int XW = (CD < 32) ? 32 : CD;        // score K (padded)
  constexpr int FXW = (DH < 32) ? 32 : DH;       // bb2 K (padded)
  constexpr int FXS = PHASE2 ? FXW : 2;
  constexpr int NPH = (DH < 16) ? 16 : DH;
  constexpr int DOUTP = PHASE2 ? CD : DH;
  constexpr int FSLD = (DH < 16) ? 16 : DH;      // gather-source stride
  constexpr int OP = PHASE2 ? ((CD < 32) ? 32 : CD) : ((DH < 16) ? 16 : DH);
  constexpr bool TDM_STAGE = (CD <= 64);         // LDS budget for fcWt tile
  constexpr int FCWSZ = TDM_STAGE ? CD * XW : 8;

  __shared__ int sh_idx[WPB][16];
  __shared__ alignas(16) _Float16 sh_a10[WPB][16][32];
  __shared__ alignas(16) _Float16 sh_fx[WPB][16][FXS];
  __shared__ alignas(16) _Float16 sh_x[WPB][16][XW];
  __shared__ alignas(16) _Float16 sh_fcw[FCWSZ];     // block-shared fcWt tile
  __shared__ float sh_s[WPB][16][CD];
  __shared__ float sh_agg[WPB][CD];

  const int w = threadIdx.x >> 5;
  const int lane = threadIdx.x & 31;
  const int p = blockIdx.x * WPB + w;   // point id in [0, 2N)
  const int b = p / N;
  __builtin_prefetch(fcW, 0, 0);

  // --- stage fcWt[CD][XW] into LDS (TDM async tensor load, wave 0 only) ---
  if (TDM_STAGE) {
#if HAVE_TDM
    if (w == 0) {   // wave-uniform branch; TDM ignores EXEC
      unsigned lds_off = (unsigned)(size_t)(void*)&sh_fcw[0];
      unsigned long long ga = (unsigned long long)(size_t)fcW;
      // D# group0: count=1, lds_addr, 57-bit global addr, type=2 ("image")
      v4u g0 = {1u, lds_off, (unsigned)ga,
                (unsigned)((ga >> 32) & 0x1FFFFFFu) | (2u << 30)};
      // D# group1: data_size=1(2B); tensor_dim0=XW, tensor_dim1=CD;
      //            tile_dim0=XW, tile_dim1=CD; tensor_dim0_stride=XW
      v8i g1 = {(int)(1u << 16),
                (int)(((unsigned)XW & 0xFFFFu) << 16),
                (int)(((unsigned)CD & 0xFFFFu) << 16),
                (int)((unsigned)XW << 16),
                (int)CD,
                (int)XW,
                0, 0};
      v4i gz4 = {0, 0, 0, 0};
      v8i gz8 = {0, 0, 0, 0, 0, 0, 0, 0};
      __builtin_amdgcn_tensor_load_to_lds(g0, g1, gz4, gz4, gz8, 0);
      __builtin_amdgcn_s_wait_tensorcnt((short)0);
    }
#else
    for (int e = threadIdx.x; e < CD * XW; e += WPB * 32) sh_fcw[e] = fcW[e];
#endif
  }

  // --- stage neighbor indices + 10-dim relpos rows (f16, K-padded to 32) ---
  if (lane < 16) {
    int idx = neigh[(size_t)p * 16 + lane];
    sh_idx[w][lane] = idx;
    const float* cc = xyz + (size_t)p * 3;
    const float* nb = xyz + ((size_t)b * N + idx) * 3;
    float cx = cc[0], cy = cc[1], cz = cc[2];
    float nx = nb[0], ny = nb[1], nz = nb[2];
    float rx = cx - nx, ry = cy - ny, rz = cz - nz;
    float dist = sqrtf(rx * rx + ry * ry + rz * rz);
    _Float16* row = sh_a10[w][lane];
    row[0] = (_Float16)dist;
    row[1] = (_Float16)rx; row[2] = (_Float16)ry; row[3] = (_Float16)rz;
    row[4] = (_Float16)cx; row[5] = (_Float16)cy; row[6] = (_Float16)cz;
    row[7] = (_Float16)nx; row[8] = (_Float16)ny; row[9] = (_Float16)nz;
#pragma unroll
    for (int q = 10; q < 32; ++q) row[q] = (_Float16)0.0f;
  }
  if (CD < XW) {  // zero padding cols (layer 0 only)
    for (int e = lane; e < 16 * XW; e += 32) sh_x[w][e / XW][e % XW] = (_Float16)0.0f;
  }
  if (PHASE2 && DH < FXW) {
    for (int e = lane; e < 16 * FXS; e += 32) sh_fx[w][e / FXS][e % FXS] = (_Float16)0.0f;
  }
  __syncthreads();

  // --- f_xyz = relu(bn(relpos[16,10] @ bb1W)) ---
  {
    v16h a = lds_a(&sh_a10[w][0][0], 32, 0);
#pragma unroll 1
    for (int n0 = 0; n0 < DH; n0 += 16) {
      v8f c = wmma_f16(a, frag_bT<32>(bb1W, 0, n0), v8f{});
      int cn = n0 + (lane & 15);
      bool ok = cn < DH;
      float s = ok ? bb1g[cn] * BN_INV : 0.0f;
      float bb = ok ? bb1b[cn] : 0.0f;
      int mb = (lane >> 4) << 3;
#pragma unroll
      for (int v = 0; v < 8; ++v) {
        float y = fmaxf(c[v] * s + bb, 0.0f);
        if (ok) {
          if (PHASE2) sh_fx[w][mb + v][cn] = (_Float16)y;
          else        sh_x[w][mb + v][DH + cn] = (_Float16)y;
        }
      }
    }
  }

  // --- gather f_nb -> x[:, 0:DH] ---
  for (int e = lane; e < 16 * DH; e += 32) {
    int j = e / DH, c = e % DH;
    int idx = sh_idx[w][j];
    sh_x[w][j][c] = (_Float16)fsrc[((size_t)b * N + idx) * FSLD + c];
  }
  __syncthreads();

  if (PHASE2) {  // f_xyz2 = relu(bn(f_xyz @ bb2W)) -> x[:, DH:]
#pragma unroll 1
    for (int n0 = 0; n0 < DH; n0 += 16) {
      v8f c = {};
#pragma unroll
      for (int k0 = 0; k0 < FXS; k0 += 32)
        c = wmma_f16(lds_a(&sh_fx[w][0][0], FXS, k0), frag_bT<FXW>(bb2W, k0, n0), c);
      int cn = n0 + (lane & 15);
      bool ok = cn < DH;
      float s = ok ? bb2g[cn] * BN_INV : 0.0f;
      float bb = ok ? bb2b[cn] : 0.0f;
      int mb = (lane >> 4) << 3;
#pragma unroll
      for (int v = 0; v < 8; ++v) {
        float y = fmaxf(c[v] * s + bb, 0.0f);
        if (ok) sh_x[w][mb + v][DH + cn] = (_Float16)y;
      }
    }
    __syncthreads();
  }

  // --- scores = x[16,CD] @ fcW[CD,CD] (B from LDS when staged) ---
#pragma unroll 1
  for (int n0 = 0; n0 < CD; n0 += 16) {
    v8f c = {};
#pragma unroll
    for (int k0 = 0; k0 < XW; k0 += 32) {
      v16h a = lds_a(&sh_x[w][0][0], XW, k0);
      v16h bf;
      if (TDM_STAGE) bf = frag_bT<XW>(&sh_fcw[0], k0, n0);
      else           bf = frag_bT<XW>(fcW, k0, n0);
      c = wmma_f16(a, bf, c);
    }
    int cn = n0 + (lane & 15);
    int mb = (lane >> 4) << 3;
#pragma unroll
    for (int v = 0; v < 8; ++v) sh_s[w][mb + v][cn] = c[v];
  }
  __syncthreads();

  // --- per-column softmax over the 16 neighbors + weighted aggregation ---
  for (int c = lane; c < CD; c += 32) {
    float mx = -1e30f;
#pragma unroll
    for (int m = 0; m < 16; ++m) mx = fmaxf(mx, sh_s[w][m][c]);
    float sum = 0.0f;
#pragma unroll
    for (int m = 0; m < 16; ++m) sum += __expf(sh_s[w][m][c] - mx);
    float inv = 1.0f / sum;
    float acc = 0.0f;
#pragma unroll
    for (int m = 0; m < 16; ++m)
      acc += (float)sh_x[w][m][c] * __expf(sh_s[w][m][c] - mx) * inv;
    sh_agg[w][c] = acc;
  }
  __syncthreads();

  // --- attention MLP (GEMV) + BN + ReLU -> out[p][0:OP] (tail zeroed) ---
  for (int co = lane; co < OP; co += 32) {
    float y = 0.0f;
    if (co < DOUTP) {
      float acc = 0.0f;
      for (int c = 0; c < CD; ++c) acc += sh_agg[w][c] * mlpW[(size_t)c * DOUTP + co];
      y = fmaxf(acc * (mlpg[co] * BN_INV) + mlpb[co], 0.0f);
    }
    out[(size_t)p * OP + co] = y;
  }
}

// ---------------------------------------------------------------------------
// Utility kernels
// ---------------------------------------------------------------------------
// f32 W[K][N] -> f16 transposed padded Wt[NP][KP], zeros in pad
__global__ void cvt_pad_tr_kernel(const float* __restrict__ s, _Float16* __restrict__ d,
                                  int K, int N, int KP, int NP) {
  int t = blockIdx.x * blockDim.x + threadIdx.x;
  if (t >= KP * NP) return;
  int n = t / KP, k = t - n * KP;
  d[t] = (k < K && n < N) ? (_Float16)s[(size_t)k * N + n] : (_Float16)0.0f;
}

__global__ void subsample_kernel(const float* __restrict__ src, const int* __restrict__ sub,
                                 float* __restrict__ dst, int N, int Nsub, int C, int tot) {
  int t = blockIdx.x * blockDim.x + threadIdx.x;
  if (t >= tot) return;
  int b = t / (Nsub * C);
  int r = t - b * (Nsub * C);
  int j = r / C;
  int c = r - j * C;
  dst[t] = src[((size_t)b * N + sub[b * Nsub + j]) * C + c];
}

// ---------------------------------------------------------------------------
// Host side
// ---------------------------------------------------------------------------
static constexpr size_t WS_FEATA = 0;               // 16 MB, stride max(din,32)
static constexpr size_t WS_F1    = 16ull << 20;     //  8 MB, stride max(dh,16)
static constexpr size_t WS_FAP1  = 24ull << 20;     //  8 MB
static constexpr size_t WS_FBB   = 32ull << 20;     // 16 MB, stride max(d,32)
static constexpr size_t WS_RES   = 48ull << 20;     // 16 MB
static constexpr size_t WS_W16   = 64ull << 20;     //  4 MB f16 padded weights

template <int CD, int WPB>
static void run_bb_phases(hipStream_t stream, int N,
                          const float* xyz, const int* neigh,
                          const float* f1, float* fap1, float* fbb,
                          const _Float16* bb1W, const float* bb1g, const float* bb1b,
                          const _Float16* bb2W, const float* bb2g, const float* bb2b,
                          const _Float16* fc1W, const float* ap1mW,
                          const float* ap1mg, const float* ap1mb,
                          const _Float16* fc2W, const float* ap2mW,
                          const float* ap2mg, const float* ap2mb) {
  int P = 2 * N;
  bb_kernel<CD, false, WPB><<<P / WPB, WPB * 32, 0, stream>>>(
      xyz, neigh, f1, bb1W, bb1g, bb1b, bb2W, bb2g, bb2b, fc1W,
      ap1mW, ap1mg, ap1mb, fap1, N);
  bb_kernel<CD, true, WPB><<<P / WPB, WPB * 32, 0, stream>>>(
      xyz, neigh, fap1, bb1W, bb1g, bb1b, bb2W, bb2g, bb2b, fc2W,
      ap2mW, ap2mg, ap2mb, fbb, N);
}

extern "C" void kernel_launch(void* const* d_in, const int* in_sizes, int n_in,
                              void* d_out, int out_size, void* d_ws, size_t ws_size,
                              hipStream_t stream) {
  (void)in_sizes; (void)n_in; (void)out_size; (void)ws_size;
  const int Ns[4]  = {65536, 16384, 4096, 1024};
  const int Din[4] = {8, 32, 128, 256};
  const int Dd[4]  = {16, 64, 128, 256};  // d = DOUT/2

  const float* features = (const float*)d_in[0];
  const float* xyz[4]; const int* neigh[4]; const int* sub[4];
  for (int i = 0; i < 4; ++i) {
    xyz[i]   = (const float*)d_in[1 + 3 * i];
    neigh[i] = (const int*)d_in[2 + 3 * i];
    sub[i]   = (const int*)d_in[3 + 3 * i];
  }
  const float* fc0W = (const float*)d_in[13];
  const float* fc0b = (const float*)d_in[14];
  const float* fc0g = (const float*)d_in[15];
  // per-layer leaves (jax sorted-key order), 23 per layer:
  // 0 ap1.fc_W 1 ap1.mlp.W 2 ap1.mlp.b 3 ap1.mlp.g 4 ap2.fc_W 5 ap2.mlp.W
  // 6 ap2.mlp.b 7 ap2.mlp.g 8 bb1.W 9 bb1.b 10 bb1.g 11 bb2.W 12 bb2.b 13 bb2.g
  // 14 mlp1.W 15 mlp1.b 16 mlp1.g 17 mlp2.W 18 mlp2.b 19 mlp2.g 20 sc.W 21 sc.b 22 sc.g
  auto PL = [&](int L, int o) { return (const float*)d_in[16 + L * 23 + o]; };

  char* ws = (char*)d_ws;
  float* featA = (float*)(ws + WS_FEATA);
  float* f1    = (float*)(ws + WS_F1);
  float* fap1  = (float*)(ws + WS_FAP1);
  float* fbb   = (float*)(ws + WS_FBB);
  float* res   = (float*)(ws + WS_RES);
  _Float16* w16 = (_Float16*)(ws + WS_W16);

  // --- convert + zero-pad + transpose weight matrices to f16 ---
  size_t woff = 0;
  _Float16 *m1W16[4], *bb1W16[4], *bb2W16[4], *fc1W16[4], *fc2W16[4], *m2W16[4], *scW16[4];
  auto cvt = [&](const float* src, int K, int N, int KP, int NP) -> _Float16* {
    _Float16* dst = w16 + woff;
    size_t cnt = (size_t)KP * NP;
    woff += (cnt + 7) & ~(size_t)7;
    cvt_pad_tr_kernel<<<(int)((cnt + 255) / 256), 256, 0, stream>>>(src, dst, K, N, KP, NP);
    return dst;
  };
  for (int L = 0; L < 4; ++L) {
    int din = Din[L], d = Dd[L], dh = d / 2;
    int DINP = din < 32 ? 32 : din;
    int DP = d < 32 ? 32 : d;
    int DHP32 = dh < 32 ? 32 : dh;
    int NPH = dh < 16 ? 16 : dh;
    m1W16[L]  = cvt(PL(L, 14), din, dh, DINP, NPH);
    bb1W16[L] = cvt(PL(L, 8), 10, dh, 32, NPH);
    bb2W16[L] = cvt(PL(L, 11), dh, dh, DHP32, NPH);
    fc1W16[L] = cvt(PL(L, 0), d, d, DP, d);
    fc2W16[L] = cvt(PL(L, 4), d, d, DP, d);
    m2W16[L]  = cvt(PL(L, 17), d, 2 * d, DP, 2 * d);
    scW16[L]  = cvt(PL(L, 20), din, 2 * d, DINP, 2 * d);
  }

  // --- fc0 ---
  {
    int M = 2 * Ns[0];
    fc0_kernel<<<(M + 255) / 256, 256, 0, stream>>>(features, fc0W, fc0g, fc0b, featA, M);
  }

  for (int L = 0; L < 4; ++L) {
    int N = Ns[L], d = Dd[L], M = 2 * N;

    // mlp1: featA[M, DINP] -> f1[M, NPH]
    switch (L) {
      case 0: convbn_kernel<32, 8, 16><<<dim3(M / 16, 1), 32, 0, stream>>>(
                featA, m1W16[L], PL(L, 16), PL(L, 15), f1); break;
      case 1: convbn_kernel<32, 32, 32><<<dim3(M / 16, 2), 32, 0, stream>>>(
                featA, m1W16[L], PL(L, 16), PL(L, 15), f1); break;
      case 2: convbn_kernel<128, 64, 64><<<dim3(M / 16, 4), 32, 0, stream>>>(
                featA, m1W16[L], PL(L, 16), PL(L, 15), f1); break;
      default: convbn_kernel<256, 128, 128><<<dim3(M / 16, 8), 32, 0, stream>>>(
                featA, m1W16[L], PL(L, 16), PL(L, 15), f1); break;
    }

    // building block (two attention-pool phases, WMMA-heavy)
    switch (L) {
      case 0: run_bb_phases<16, 4>(stream, N, xyz[L], neigh[L], f1, fap1, fbb,
                bb1W16[L], PL(L, 10), PL(L, 9), bb2W16[L], PL(L, 13), PL(L, 12),
                fc1W16[L], PL(L, 1), PL(L, 3), PL(L, 2),
                fc2W16[L], PL(L, 5), PL(L, 7), PL(L, 6)); break;
      case 1: run_bb_phases<64, 4>(stream, N, xyz[L], neigh[L], f1, fap1, fbb,
                bb1W16[L], PL(L, 10), PL(L, 9), bb2W16[L], PL(L, 13), PL(L, 12),
                fc1W16[L], PL(L, 1), PL(L, 3), PL(L, 2),
                fc2W16[L], PL(L, 5), PL(L, 7), PL(L, 6)); break;
      case 2: run_bb_phases<128, 4>(stream, N, xyz[L], neigh[L], f1, fap1, fbb,
                bb1W16[L], PL(L, 10), PL(L, 9), bb2W16[L], PL(L, 13), PL(L, 12),
                fc1W16[L], PL(L, 1), PL(L, 3), PL(L, 2),
                fc2W16[L], PL(L, 5), PL(L, 7), PL(L, 6)); break;
      default: run_bb_phases<256, 2>(stream, N, xyz[L], neigh[L], f1, fap1, fbb,
                bb1W16[L], PL(L, 10), PL(L, 9), bb2W16[L], PL(L, 13), PL(L, 12),
                fc1W16[L], PL(L, 1), PL(L, 3), PL(L, 2),
                fc2W16[L], PL(L, 5), PL(L, 7), PL(L, 6)); break;
    }

    // fused mlp2 + shortcut + leaky_relu -> res[M, 2d]
    switch (L) {
      case 0: resout_kernel<32, 32, 32><<<dim3(M / 16, 2), 32, 0, stream>>>(
                fbb, m2W16[L], PL(L, 19), PL(L, 18),
                featA, scW16[L], PL(L, 22), PL(L, 21), res); break;
      case 1: resout_kernel<64, 32, 128><<<dim3(M / 16, 8), 32, 0, stream>>>(
                fbb, m2W16[L], PL(L, 19), PL(L, 18),
                featA, scW16[L], PL(L, 22), PL(L, 21), res); break;
      case 2: resout_kernel<128, 128, 256><<<dim3(M / 16, 16), 32, 0, stream>>>(
                fbb, m2W16[L], PL(L, 19), PL(L, 18),
                featA, scW16[L], PL(L, 22), PL(L, 21), res); break;
      default: resout_kernel<256, 256, 512><<<dim3(M / 16, 32), 32, 0, stream>>>(
                fbb, m2W16[L], PL(L, 19), PL(L, 18),
                featA, scW16[L], PL(L, 22), PL(L, 21), res); break;
    }

    // subsample -> next layer input (C = 2d = next din, already mult of 32)
    int Nsub = N / 4, C = 2 * d;
    float* dst = (L == 3) ? (float*)d_out : featA;
    int tot = 2 * Nsub * C;
    subsample_kernel<<<(tot + 255) / 256, 256, 0, stream>>>(res, sub[L], dst, N, Nsub, C, tot);
  }
}
